// backbone_9792525435444
// MI455X (gfx1250) — matrix operand order repaired and checked by reading, not compiled, not run on hardware
//
#include <hip/hip_runtime.h>
#include <hip/hip_bf16.h>
#include <stddef.h>
#include <stdint.h>

// ---------------------------------------------------------------------------
// CDNA5 (gfx1250, wave32) CondConv shuffle block.
//
//   1) attn_kernel          : pooled mean -> sigmoid attention  [3][32][4]
//   2) wagg_kernel<KH,KW>   : per-sample expert-aggregated bf16 weights,
//                             stored in WMMA B-fragment order [kstep][oc][kk]
//   3) condconv_wmma<...>   : implicit-GEMM conv; weights DMA'd into LDS via
//                             tensor_load_to_lds (TDM), input tile staged as
//                             [row][col][ch] so every A/B fragment is two
//                             contiguous ds_load_b128; v_wmma_f32_16x16x32_bf16
//   4) bnstats_kernel       : batch mean/var -> (scale, shift)
//   5) shuffle_kernel       : BN apply + concat + channel_shuffle(g=8)
// ---------------------------------------------------------------------------

typedef __attribute__((ext_vector_type(16))) __bf16 v16bf;
typedef __attribute__((ext_vector_type(8)))  float  v8f;
typedef __attribute__((ext_vector_type(4)))  unsigned int u32x4;
typedef __attribute__((ext_vector_type(8)))  int i32x8;
typedef __attribute__((ext_vector_type(4)))  int i32x4;

union V16U { v16bf v; uint4 q[2]; unsigned short u[16]; };

#if defined(__HIP_DEVICE_COMPILE__) && __has_builtin(__builtin_amdgcn_tensor_load_to_lds)
#define USE_TDM 1
#endif

__device__ __forceinline__ unsigned short f2bf(float f) {
    unsigned u = __float_as_uint(f);
    unsigned r = u + 0x7FFFu + ((u >> 16) & 1u);   // round-to-nearest-even
    return (unsigned short)(r >> 16);
}

// ---------------------------------------------------------------------------
// 1) Attention: grid = 32 (samples), block = 256
// ---------------------------------------------------------------------------
__global__ void attn_kernel(const float* __restrict__ x, int cbase,
                            const float* __restrict__ aw, const float* __restrict__ ab,
                            float* __restrict__ att_out) {
    __shared__ float part[256];
    __shared__ float pooled[64];
    const int n = blockIdx.x;
    const int t = threadIdx.x;
    const int c = t >> 2;
    const int q = t & 3;

    const float* px = x + (((size_t)n * 256 + cbase + c) * 4096) + q * 1024;
    float s = 0.f;
    for (int p = 0; p < 1024; p += 4) {
        float4 v = *(const float4*)(px + p);
        s += v.x + v.y + v.z + v.w;
    }
    part[t] = s;
    __syncthreads();
    if (t < 64)
        pooled[t] = (part[4 * t] + part[4 * t + 1] + part[4 * t + 2] + part[4 * t + 3])
                    * (1.0f / 4096.0f);
    __syncthreads();
    if (t < 4) {
        float z = ab[t];
        for (int cc = 0; cc < 64; ++cc) z += pooled[cc] * aw[t * 64 + cc];
        att_out[n * 4 + t] = 1.0f / (1.0f + __expf(-z));
    }
}

// ---------------------------------------------------------------------------
// 2) Per-sample expert aggregation into bf16, B-fragment order.
//    k' = tap*64 + i  (tap = y*KW + x);  out[idx], idx = (kstep*64 + oc)*32 + kk
//    grid = 32 (samples), block = 256
// ---------------------------------------------------------------------------
template <int KH, int KW>
__global__ void wagg_kernel(const float* __restrict__ w,    // [4][64][64][KH][KW]
                            const float* __restrict__ att,  // [32][4]
                            unsigned short* __restrict__ wagg) {
    constexpr int KHKW = KH * KW;
    constexpr int KRED = 64 * KHKW;
    __shared__ float a_s[4];
    const int n = blockIdx.x;
    if (threadIdx.x < 4) a_s[threadIdx.x] = att[n * 4 + threadIdx.x];
    __syncthreads();
    const float a0 = a_s[0], a1 = a_s[1], a2 = a_s[2], a3 = a_s[3];

    unsigned short* outp = wagg + (size_t)n * (KRED * 64);
    constexpr size_t ESTR = (size_t)64 * 64 * KHKW;
    for (int idx = threadIdx.x; idx < KRED * 64; idx += 256) {
        const int kk = idx & 31;
        const int o  = (idx >> 5) & 63;
        const int s  = idx >> 11;
        const int kp = s * 32 + kk;          // k' in tap-major order
        const int tap = kp >> 6;
        const int i   = kp & 63;
        const size_t g = (size_t)(o * 64 + i) * KHKW + tap;
        float v = a0 * w[g] + a1 * w[g + ESTR] + a2 * w[g + 2 * ESTR] + a3 * w[g + 3 * ESTR];
        outp[idx] = f2bf(v);
    }
}

// ---------------------------------------------------------------------------
// 3) CondConv as implicit GEMM.
//    Block = sample x 2 output rows (M=128 px, N=64 oc); 8 waves; wave = 16x64
//    via 4x v_wmma_f32_16x16x32_bf16 per K-step.
//    LDS: weights [KSTEPS][64][32] bf16 (TDM-loaded), input [4row][66col][64ch].
//    grid = dim3(32 row-tiles, 32 samples), block = 256, dynamic LDS
// ---------------------------------------------------------------------------
template <int KH, int KW, int PH, int PW>
__global__ void condconv_wmma(const float* __restrict__ x, int cbase,
                              const unsigned short* __restrict__ wagg,
                              float* __restrict__ out) {
    constexpr int KHKW   = KH * KW;
    constexpr int KRED   = 64 * KHKW;       // 576 or 192
    constexpr int KSTEPS = KRED / 32;       // 18 or 6
    constexpr int ROWOFF = 1 - PH;

    extern __shared__ unsigned short smem_u16[];
    unsigned short* lds_w  = smem_u16;                 // [KSTEPS][64][32]
    unsigned short* lds_in = lds_w + KRED * 64;        // [4][66][64]

    const int tid = threadIdx.x;
    const int n   = blockIdx.y;
    const int h0  = blockIdx.x * 2;
    const unsigned short* wsrc = wagg + (size_t)n * (KRED * 64);

    // ---- async DMA of this sample's aggregated weights into LDS (wave 0)
#if defined(USE_TDM)
    if (tid < 32) {
        constexpr unsigned NQ = (unsigned)KRED * 64 * 2 / 8;   // quadwords
        const uint64_t ga = (uint64_t)(uintptr_t)wsrc;
        const uint32_t la = (uint32_t)(uintptr_t)lds_w;
        u32x4 g0 = (u32x4){1u,                       // count=1, user mode
                           la,                       // lds_addr
                           (uint32_t)ga,             // global_addr[31:0]
                           (uint32_t)(ga >> 32) | (2u << 30)};  // addr hi | type=2
        i32x8 g1 = (i32x8){(int)(3u << 16),          // data_size = 8B
                           (int)(NQ << 16),          // tensor_dim0[15:0]
                           (int)((NQ >> 16) | (1u << 16)),  // dim0 hi | tensor_dim1=1
                           (int)(NQ << 16),          // tile_dim0
                           1,                        // tile_dim1 = 1
                           (int)NQ,                  // tensor_dim0_stride
                           0, 0};
        i32x4 g2 = (i32x4){0, 0, 0, 0};
        i32x4 g3 = (i32x4){0, 0, 0, 0};
        i32x8 g4 = (i32x8){0, 0, 0, 0, 0, 0, 0, 0};
        __builtin_amdgcn_tensor_load_to_lds(g0, g1, g2, g3, g4, 0);
    }
#else
    for (int idx = tid; idx < KRED * 64 / 8; idx += 256)
        ((uint4*)lds_w)[idx] = ((const uint4*)wsrc)[idx];
#endif

    // ---- stage haloed input tile (rows h0-1..h0+2), layout [row][col][ch]
    {   // zero column pads (col 0 and 65) for all (row, ch)
        const int r = tid >> 6, ch = tid & 63;
        lds_in[(r * 66 + 0)  * 64 + ch] = 0;
        lds_in[(r * 66 + 65) * 64 + ch] = 0;
    }
    for (int idx = tid; idx < 64 * 4 * 16; idx += 256) {
        const int ch = idx >> 6;
        const int r  = (idx >> 4) & 3;
        const int c4 = idx & 15;
        const int h  = h0 - 1 + r;
        float4 v = make_float4(0.f, 0.f, 0.f, 0.f);
        if ((unsigned)h < 64u)
            v = *(const float4*)(x + (((size_t)n * 256 + cbase + ch) * 64 + h) * 64 + c4 * 4);
        unsigned short* d = lds_in + (r * 66 + 1 + c4 * 4) * 64 + ch;
        d[0]   = f2bf(v.x);
        d[64]  = f2bf(v.y);
        d[128] = f2bf(v.z);
        d[192] = f2bf(v.w);
    }
#if defined(USE_TDM)
    if (tid < 32) __builtin_amdgcn_s_wait_tensorcnt(0);
#endif
    __syncthreads();

    // ---- WMMA main loop
    const int wv   = tid >> 5;
    const int lane = tid & 31;
    const int pix  = wv * 16 + (lane & 15);     // pixel within 128-px tile
    const int row_local = pix >> 6;
    const int col       = pix & 63;
    const int hiK  = (lane < 16) ? 0 : 8;       // A lane-group channel bias
    const int kb16 = (lane < 16) ? 0 : 8;       // B lane-group K bias
    const int nb   = lane & 15;

    v8f acc[4];
    #pragma unroll
    for (int t = 0; t < 4; ++t) acc[t] = (v8f){0.f,0.f,0.f,0.f,0.f,0.f,0.f,0.f};

    #pragma unroll 2
    for (int s = 0; s < KSTEPS; ++s) {
        const int tap = s >> 1;                 // 2 chunks of 32 ch per tap
        const int i0  = (s & 1) * 32;
        const int y   = tap / KW;
        const int xk  = tap - y * KW;

        // A fragment: two contiguous b128 LDS loads (channels i0+hiK.. , +16..)
        const int abase = ((row_local + y + ROWOFF) * 66 + 1 + col + xk - PW) * 64
                          + i0 + hiK;
        V16U a;
        a.q[0] = *(const uint4*)(lds_in + abase);
        a.q[1] = *(const uint4*)(lds_in + abase + 16);

        #pragma unroll
        for (int t = 0; t < 4; ++t) {
            const int oc = t * 16 + nb;
            const unsigned short* bp = lds_w + ((s * 64 + oc) * 32 + kb16);
            V16U b;
            b.q[0] = *(const uint4*)bp;
            b.q[1] = *(const uint4*)(bp + 16);
#if defined(__HIP_DEVICE_COMPILE__)
            acc[t] = __builtin_amdgcn_wmma_f32_16x16x32_bf16(
                         false, a.v, false, b.v, (short)0, acc[t], false, false);
#endif
        }
    }

    // ---- write conv result (C/D layout: VGPR j -> M=j+mhi, N=t*16+nb)
    const int mhi = (lane < 16) ? 0 : 8;
    const int p0  = h0 * 64 + wv * 16 + mhi;
    #pragma unroll
    for (int t = 0; t < 4; ++t) {
        const int o = t * 16 + nb;
        float* op = out + ((size_t)n * 64 + o) * 4096 + p0;
        *(float4*)(op)     = make_float4(acc[t][0], acc[t][1], acc[t][2], acc[t][3]);
        *(float4*)(op + 4) = make_float4(acc[t][4], acc[t][5], acc[t][6], acc[t][7]);
    }
}

// ---------------------------------------------------------------------------
// 4) BN batch statistics: grid = 64 (channels), block = 256
// ---------------------------------------------------------------------------
__global__ void bnstats_kernel(const float* __restrict__ conv,
                               const float* __restrict__ g, const float* __restrict__ b,
                               float* __restrict__ bn_out) {
    __shared__ float s1[256], s2[256];
    const int o = blockIdx.x, t = threadIdx.x;
    float su = 0.f, sq = 0.f;
    for (int n = 0; n < 32; ++n) {
        const float* p = conv + ((size_t)n * 64 + o) * 4096;
        for (int i = t * 4; i < 4096; i += 1024) {
            float4 v = *(const float4*)(p + i);
            su += v.x + v.y + v.z + v.w;
            sq += v.x * v.x + v.y * v.y + v.z * v.z + v.w * v.w;
        }
    }
    s1[t] = su; s2[t] = sq;
    __syncthreads();
    for (int st = 128; st > 0; st >>= 1) {
        if (t < st) { s1[t] += s1[t + st]; s2[t] += s2[t + st]; }
        __syncthreads();
    }
    if (t == 0) {
        const float inv  = 1.0f / (32.0f * 4096.0f);
        const float mean = s1[0] * inv;
        const float var  = s2[0] * inv - mean * mean;
        const float sc   = g[o] * rsqrtf(var + 1e-5f);
        bn_out[o * 2]     = sc;
        bn_out[o * 2 + 1] = b[o] - mean * sc;
    }
}

// ---------------------------------------------------------------------------
// 5) BN apply + concat + channel_shuffle(g=8): c_in = (c_out%8)*32 + c_out/8
//    grid = 32768, block = 256, float4 per thread
// ---------------------------------------------------------------------------
__global__ void shuffle_kernel(const float* __restrict__ x,
                               const float* __restrict__ conv_all,  // [3][32][64][4096]
                               const float* __restrict__ bn_all,    // [3][64][2]
                               float* __restrict__ out) {
    const int idx   = blockIdx.x * 256 + threadIdx.x;
    const int p4    = idx & 1023;
    const int c_out = (idx >> 10) & 255;
    const int n     = idx >> 18;
    const int c_in  = (c_out & 7) * 32 + (c_out >> 3);

    float4 v;
    if (c_in < 64) {
        v = *(const float4*)(x + ((size_t)n * 256 + c_in) * 4096 + p4 * 4);
    } else {
        const int br = (c_in >> 6) - 1;
        const int o  = c_in & 63;
        v = *(const float4*)(conv_all + (((size_t)br * 32 + n) * 64 + o) * 4096 + p4 * 4);
        const float sc = bn_all[br * 128 + o * 2];
        const float sh = bn_all[br * 128 + o * 2 + 1];
        v.x = v.x * sc + sh; v.y = v.y * sc + sh;
        v.z = v.z * sc + sh; v.w = v.w * sc + sh;
    }
    *(float4*)(out + ((size_t)n * 256 + c_out) * 4096 + p4 * 4) = v;
}

// ---------------------------------------------------------------------------
// Launcher.  Workspace (floats): att[3][32][4] @0, bn[3][64][2] @384,
// conv[3][32][64][4096] @1024, then bf16 wagg blobs (sq, ver, hor).
// ---------------------------------------------------------------------------
extern "C" void kernel_launch(void* const* d_in, const int* in_sizes, int n_in,
                              void* d_out, int out_size, void* d_ws, size_t ws_size,
                              hipStream_t stream) {
    (void)in_sizes; (void)n_in; (void)out_size; (void)ws_size;

    const float* x = (const float*)d_in[0];
    float* ws       = (float*)d_ws;
    float* att_ws   = ws;            // 3*128
    float* bn_ws    = ws + 384;      // 3*128
    float* conv_all = ws + 1024;     // 3*32*64*4096
    const size_t CONV_B = (size_t)32 * 64 * 4096;

    unsigned short* wagg_base = (unsigned short*)(ws + 1024 + 3 * CONV_B);
    unsigned short* wagg_sq  = wagg_base;                               // 32*576*64
    unsigned short* wagg_ver = wagg_sq  + (size_t)32 * 576 * 64;        // 32*192*64
    unsigned short* wagg_hor = wagg_ver + (size_t)32 * 192 * 64;        // 32*192*64

    const int cbase[3] = {64, 128, 192};
    const float* aw[3] = {(const float*)d_in[1], (const float*)d_in[6],  (const float*)d_in[11]};
    const float* ab[3] = {(const float*)d_in[2], (const float*)d_in[7],  (const float*)d_in[12]};
    const float* wt[3] = {(const float*)d_in[3], (const float*)d_in[8],  (const float*)d_in[13]};
    const float* gm[3] = {(const float*)d_in[4], (const float*)d_in[9],  (const float*)d_in[14]};
    const float* bt[3] = {(const float*)d_in[5], (const float*)d_in[10], (const float*)d_in[15]};

    // 1) attention per branch
    for (int b = 0; b < 3; ++b)
        attn_kernel<<<32, 256, 0, stream>>>(x, cbase[b], aw[b], ab[b], att_ws + b * 128);

    // 2) per-sample aggregated bf16 weights (B-fragment order)
    wagg_kernel<3, 3><<<32, 256, 0, stream>>>(wt[0], att_ws + 0 * 128, wagg_sq);
    wagg_kernel<3, 1><<<32, 256, 0, stream>>>(wt[1], att_ws + 1 * 128, wagg_ver);
    wagg_kernel<1, 3><<<32, 256, 0, stream>>>(wt[2], att_ws + 2 * 128, wagg_hor);

    // 3) CondConv branches
    const dim3 cgrid(32, 32);
    const size_t in_lds  = (size_t)4 * 66 * 64 * 2;
    const size_t sh_sq   = (size_t)576 * 64 * 2 + in_lds;   // ~105 KB (< 320 KB/WGP)
    const size_t sh_thin = (size_t)192 * 64 * 2 + in_lds;   // ~58 KB

    condconv_wmma<3, 3, 1, 1><<<cgrid, 256, sh_sq, stream>>>(
        x, cbase[0], wagg_sq, conv_all + 0 * CONV_B);
    condconv_wmma<3, 1, 1, 0><<<cgrid, 256, sh_thin, stream>>>(
        x, cbase[1], wagg_ver, conv_all + 1 * CONV_B);
    condconv_wmma<1, 3, 0, 1><<<cgrid, 256, sh_thin, stream>>>(
        x, cbase[2], wagg_hor, conv_all + 2 * CONV_B);

    // 4) BN batch stats per branch
    for (int b = 0; b < 3; ++b)
        bnstats_kernel<<<64, 256, 0, stream>>>(conv_all + b * CONV_B, gm[b], bt[b],
                                               bn_ws + b * 128);

    // 5) BN apply + concat + channel shuffle
    shuffle_kernel<<<32768, 256, 0, stream>>>(x, conv_all, bn_ws, (float*)d_out);
}
